// FeaturePropagation_16930761080949
// MI455X (gfx1250) — matrix operand order repaired and checked by reading, not compiled, not run on hardware
//
#include <hip/hip_runtime.h>
#include <hip/hip_bf16.h>
#include <math.h>

typedef float v2f  __attribute__((ext_vector_type(2)));
typedef float v8f  __attribute__((ext_vector_type(8)));
typedef __bf16 v16bf __attribute__((ext_vector_type(16)));

#define B_   4
#define N1_  8192
#define N2_  2048
#define C1_  256
#define C2_  512
#define CIN_ 768      // K
#define OUT_ 512      // M
#define CINDW_ 384    // K in dwords (bf16 pairs)

union FragU { v16bf v; uint4 q[2]; };

static __device__ __forceinline__ void split_bf16(float x, __hip_bfloat16& h, __hip_bfloat16& l) {
    h = __float2bfloat16(x);
    l = __float2bfloat16(x - __bfloat162float(h));
}

// ---------------------------------------------------------------------------
// Kernel 1: 3-NN + inverse-distance weights. xyz2[b] (24KB) staged in LDS,
// one thread per query point keeps a running top-3 (scan order matches
// lax.top_k tie-breaking).
// ---------------------------------------------------------------------------
__global__ void knn_weights_kernel(const float* __restrict__ xyz1,
                                   const float* __restrict__ xyz2,
                                   int* __restrict__ gidx,
                                   float* __restrict__ gw) {
    __shared__ float sx[N2_], sy[N2_], sz[N2_];
    const int b = blockIdx.x >> 5;              // 32 blocks per batch
    const int nbase = (blockIdx.x & 31) * 256;
    const int t = threadIdx.x;

    const float* p2 = xyz2 + (size_t)b * N2_ * 3;
    for (int i = t; i < N2_; i += 256) {
        sx[i] = p2[i * 3 + 0]; sy[i] = p2[i * 3 + 1]; sz[i] = p2[i * 3 + 2];
    }
    __syncthreads();

    const int n = nbase + t;
    const float* q = xyz1 + ((size_t)b * N1_ + n) * 3;
    const float qx = q[0], qy = q[1], qz = q[2];

    float d0 = 3.4e38f, d1 = 3.4e38f, d2 = 3.4e38f;
    int   i0 = 0, i1 = 0, i2 = 0;
    for (int j = 0; j < N2_; ++j) {
        const float dx = qx - sx[j], dy = qy - sy[j], dz = qz - sz[j];
        const float d = dx * dx + dy * dy + dz * dz;
        if (d < d0)      { d2 = d1; i2 = i1; d1 = d0; i1 = i0; d0 = d; i0 = j; }
        else if (d < d1) { d2 = d1; i2 = i1; d1 = d;  i1 = j; }
        else if (d < d2) { d2 = d;  i2 = j; }
    }
    const float e0 = sqrtf(fmaxf(d0, 1e-12f)) + 1e-8f;
    const float e1 = sqrtf(fmaxf(d1, 1e-12f)) + 1e-8f;
    const float e2 = sqrtf(fmaxf(d2, 1e-12f)) + 1e-8f;
    float w0 = 1.0f / e0, w1 = 1.0f / e1, w2 = 1.0f / e2;
    const float inv = 1.0f / (w0 + w1 + w2);
    w0 *= inv; w1 *= inv; w2 *= inv;

    const size_t base = ((size_t)b * N1_ + n) * 3;
    gidx[base + 0] = i0; gidx[base + 1] = i1; gidx[base + 2] = i2;
    gw[base + 0] = w0;  gw[base + 1] = w1;  gw[base + 2] = w2;
}

// ---------------------------------------------------------------------------
// Kernel 2: split W (512x768 f32) into bf16 hi/lo, row-major.
// ---------------------------------------------------------------------------
__global__ void split_w_kernel(const float* __restrict__ Wf,
                               __hip_bfloat16* __restrict__ Whi,
                               __hip_bfloat16* __restrict__ Wlo) {
    const int i = blockIdx.x * 256 + threadIdx.x;   // exactly OUT_*CIN_ threads
    const float v = Wf[i];
    __hip_bfloat16 h, l; split_bf16(v, h, l);
    Whi[i] = h; Wlo[i] = l;
}

// ---------------------------------------------------------------------------
// Kernel 3: build X^T = [interp(feats2); feats1] as (B, N1, 768) bf16 hi/lo.
// Transposed layout => GEMM B-fragments are contiguous dword runs.
// Block: 64 n of one batch; thread -> (n = t>>2, 4-channel lane = t&3).
// ---------------------------------------------------------------------------
__global__ void build_xt_kernel(const float* __restrict__ feats1,
                                const float* __restrict__ feats2,
                                const int* __restrict__ gidx,
                                const float* __restrict__ gw,
                                __hip_bfloat16* __restrict__ Xhi,
                                __hip_bfloat16* __restrict__ Xlo) {
    __shared__ int   sidx[64 * 3];
    __shared__ float sw[64 * 3];
    const int b = blockIdx.x >> 7;              // 128 blocks per batch
    const int nbase = (blockIdx.x & 127) * 64;
    const int t = threadIdx.x;

    if (t < 192) {
        const int n = nbase + t / 3, k = t % 3;
        sidx[t] = gidx[((size_t)b * N1_ + n) * 3 + k];
        sw[t]   = gw[((size_t)b * N1_ + n) * 3 + k];
    }
    __syncthreads();

    const int nl = t >> 2, cq = t & 3;
    const int n = nbase + nl;
    const int   i0 = sidx[nl * 3], i1 = sidx[nl * 3 + 1], i2 = sidx[nl * 3 + 2];
    const float w0 = sw[nl * 3],   w1 = sw[nl * 3 + 1],   w2 = sw[nl * 3 + 2];

    __hip_bfloat16* ph = Xhi + ((size_t)b * N1_ + n) * CIN_;
    __hip_bfloat16* pl = Xlo + ((size_t)b * N1_ + n) * CIN_;

    for (int c0 = cq * 4; c0 < CIN_; c0 += 16) {
        float v[4];
        if (c0 < C2_) {
            for (int u = 0; u < 4; ++u) {
                const float* row = feats2 + ((size_t)b * C2_ + c0 + u) * N2_;  // L2-resident gathers
                v[u] = w0 * row[i0] + w1 * row[i1] + w2 * row[i2];
            }
        } else {
            for (int u = 0; u < 4; ++u)
                v[u] = feats1[((size_t)b * C1_ + (c0 + u - C2_)) * N1_ + n];
        }
        union { __hip_bfloat16 h[4]; uint2 u2; } hh, lo;
        for (int u = 0; u < 4; ++u) split_bf16(v[u], hh.h[u], lo.h[u]);
        *(uint2*)(ph + c0) = hh.u2;   // 8B aligned: row stride 1536B, c0 mult of 4
        *(uint2*)(pl + c0) = lo.u2;
    }
}

// ---------------------------------------------------------------------------
// Kernel 4: Y[b] = W * X[b] via 3-pass split-bf16 V_WMMA_F32_16X16X32_BF16.
// Block = 512 thr (16 wave32), tile 128(M) x 128(N), KC = 64 (2 wmma K-steps).
// Per-wave work is only 2x2 16x16 subtiles -> 4 f32 accumulators (32 VGPRs),
// keeping the live set ~100 VGPRs so the WMMA loop stays spill-free.
// LDS holds tiles in FRAGMENT-NATIVE layout: per fragment, lane*8 dwords
// contiguous -> fragment read = two ds_load_b128, no operand repacking.
//   A frag dword (lane,v): M=lane%16, Kdw = kstep*16 + (lane/16)*4 + (v&3) + (v>=4)*8
//   B frag dword (lane,v): N=lane%16, Kdw = kstep*16 + (lane/16)*4 + (v&3) + (v>=4)*8
// (ISA 7.12.2 16-bit A 16x32 / B 32x16 wave32 layouts.)
// ---------------------------------------------------------------------------
#define BM 128
#define BN 128
#define FRAG_DW 256        // 32 lanes * 8 dwords per fragment

__global__ void __launch_bounds__(512, 1)
gemm_wmma_bf16_kernel(const uint32_t* __restrict__ WhiDw,
                      const uint32_t* __restrict__ WloDw,
                      const uint32_t* __restrict__ XhiDw,
                      const uint32_t* __restrict__ XloDw,
                      float* __restrict__ Y) {
    __shared__ uint32_t sA[32 * FRAG_DW];   // 8 mt x 2 kstep x 2 part (32 KB)
    __shared__ uint32_t sB[32 * FRAG_DW];   // 8 nt x 2 kstep x 2 part (32 KB)

    const int t = threadIdx.x;
    const int nbase = blockIdx.x * BN;
    const int obase = blockIdx.y * BM;
    const int b = blockIdx.z;
    const uint32_t* Xh = XhiDw + (size_t)b * N1_ * CINDW_;
    const uint32_t* Xl = XloDw + (size_t)b * N1_ * CINDW_;

    const int lane = t & 31, ll = lane & 15, lh = lane >> 4;
    const int wv = t >> 5;                       // 16 waves
    const int wm = wv & 3, wn = wv >> 2;         // wave: mt = wm*2+{0,1}, nt = wn*2+{0,1}

    v8f acc[2][2];
    for (int im = 0; im < 2; ++im)
        for (int jn = 0; jn < 2; ++jn)
            for (int r = 0; r < 8; ++r) acc[im][jn][r] = 0.0f;

    for (int cd = 0; cd < CINDW_; cd += 32) {    // 64 bf16 K per chunk
        // ---- stage A & B tiles as fragments: 4096 b128 copies, 8/thread ----
        for (int s = t; s < 4096; s += 512) {
            const int isB   = s >> 11;
            const int r     = s & 2047;
            const int frag  = r >> 6;            // mt*4 + kstep*2 + part
            const int lane2 = (r >> 1) & 31;
            const int half  = r & 1;
            const int part  = frag & 1;
            const int kstep = (frag >> 1) & 1;
            const int mt    = frag >> 2;
            const int ll2 = lane2 & 15, lh2 = lane2 >> 4;
            const int row   = (isB ? nbase : obase) + mt * 16 + ll2;
            const int dwcol = cd + kstep * 16 +
                              (isB ? (lh2 * 4 + half * 8)      // B: K {0-7,16-23} | {8-15,24-31}
                                   : (lh2 * 4 + half * 8));    // A: K {0-7,16-23} | {8-15,24-31}
            const uint32_t* srcBase = isB ? (part ? Xl : Xh) : (part ? WloDw : WhiDw);
            const uint4 v = *(const uint4*)(srcBase + (size_t)row * CINDW_ + dwcol);
            uint32_t* dst = (isB ? sB : sA) + frag * FRAG_DW + lane2 * 8 + half * 4;
            *(uint4*)dst = v;
        }
        __syncthreads();

        // Prefetch next K-chunk of X^T into L2 (global_prefetch_b8).
        if (cd + 32 < CINDW_)
            __builtin_prefetch(Xh + (size_t)(nbase + (t >> 2)) * CINDW_ + cd + 32 + (t & 3) * 8, 0, 0);

#pragma unroll
        for (int kstep = 0; kstep < 2; ++kstep) {
            FragU ah[2], al[2];
#pragma unroll
            for (int im = 0; im < 2; ++im) {
                const int mt = wm * 2 + im;
                const uint32_t* pH = sA + (mt * 4 + kstep * 2 + 0) * FRAG_DW + lane * 8;
                const uint32_t* pL = sA + (mt * 4 + kstep * 2 + 1) * FRAG_DW + lane * 8;
                ah[im].q[0] = *(const uint4*)pH;       ah[im].q[1] = *(const uint4*)(pH + 4);
                al[im].q[0] = *(const uint4*)pL;       al[im].q[1] = *(const uint4*)(pL + 4);
            }
#pragma unroll
            for (int jn = 0; jn < 2; ++jn) {
                const int nt = wn * 2 + jn;
                FragU bh, bl;
                const uint32_t* pH = sB + (nt * 4 + kstep * 2 + 0) * FRAG_DW + lane * 8;
                const uint32_t* pL = sB + (nt * 4 + kstep * 2 + 1) * FRAG_DW + lane * 8;
                bh.q[0] = *(const uint4*)pH;           bh.q[1] = *(const uint4*)(pH + 4);
                bl.q[0] = *(const uint4*)pL;           bl.q[1] = *(const uint4*)(pL + 4);
#pragma unroll
                for (int im = 0; im < 2; ++im) {
                    acc[im][jn] = __builtin_amdgcn_wmma_f32_16x16x32_bf16(
                        false, ah[im].v, false, bh.v, (short)0, acc[im][jn], false, false);
                    acc[im][jn] = __builtin_amdgcn_wmma_f32_16x16x32_bf16(
                        false, ah[im].v, false, bl.v, (short)0, acc[im][jn], false, false);
                    acc[im][jn] = __builtin_amdgcn_wmma_f32_16x16x32_bf16(
                        false, al[im].v, false, bh.v, (short)0, acc[im][jn], false, false);
                }
            }
        }
        __syncthreads();
    }

    // C/D layout: VGPR r holds rows M=r (lanes 0-15) and M=r+8 (lanes 16-31).
    float* Yb = Y + ((size_t)b * OUT_ + obase) * N1_;
    for (int im = 0; im < 2; ++im) {
        for (int jn = 0; jn < 2; ++jn) {
            const int col = nbase + (wn * 2 + jn) * 16 + ll;
            for (int r = 0; r < 8; ++r) {
                const int row = (wm * 2 + im) * 16 + r + lh * 8;
                Yb[(size_t)row * N1_ + col] = acc[im][jn][r];
            }
        }
    }
}

// ---------------------------------------------------------------------------
// Kernel 5: per-channel mean / biased var over (B, N1). Deterministic tree.
// ---------------------------------------------------------------------------
__global__ void bn_stats_kernel(const float* __restrict__ Y,
                                float* __restrict__ mean,
                                float* __restrict__ var) {
    __shared__ float ss[256], sq[256];
    const int o = blockIdx.x, t = threadIdx.x;
    float s = 0.0f, q = 0.0f;
    for (int i = t; i < B_ * N1_; i += 256) {
        const int b = i >> 13, n = i & (N1_ - 1);
        const float v = Y[((size_t)b * OUT_ + o) * N1_ + n];
        s += v; q += v * v;
    }
    ss[t] = s; sq[t] = q;
    __syncthreads();
    for (int st = 128; st > 0; st >>= 1) {
        if (t < st) { ss[t] += ss[t + st]; sq[t] += sq[t + st]; }
        __syncthreads();
    }
    if (t == 0) {
        const float inv = 1.0f / (float)(B_ * N1_);
        const float m = ss[0] * inv;
        mean[o] = m;
        var[o]  = sq[0] * inv - m * m;
    }
}

// ---------------------------------------------------------------------------
// Kernel 6: BN normalize + gamma/beta + ReLU -> d_out.
// ---------------------------------------------------------------------------
__global__ void bn_apply_kernel(const float* __restrict__ Y,
                                const float* __restrict__ mean,
                                const float* __restrict__ var,
                                const float* __restrict__ gamma,
                                const float* __restrict__ beta,
                                float* __restrict__ outp) {
    const size_t total = (size_t)B_ * OUT_ * N1_;
    for (size_t i = (size_t)blockIdx.x * 256 + threadIdx.x; i < total;
         i += (size_t)gridDim.x * 256) {
        const int o = (int)((i / N1_) % OUT_);
        const float v = (Y[i] - mean[o]) * rsqrtf(var[o] + 1e-5f) * gamma[o] + beta[o];
        outp[i] = fmaxf(v, 0.0f);
    }
}

// ---------------------------------------------------------------------------
extern "C" void kernel_launch(void* const* d_in, const int* in_sizes, int n_in,
                              void* d_out, int out_size, void* d_ws, size_t ws_size,
                              hipStream_t stream) {
    const float* xyz1   = (const float*)d_in[0];
    const float* xyz2   = (const float*)d_in[1];
    const float* feats1 = (const float*)d_in[2];
    const float* feats2 = (const float*)d_in[3];
    const float* Wf     = (const float*)d_in[4];
    const float* gamma  = (const float*)d_in[5];
    const float* beta   = (const float*)d_in[6];
    float* outp = (float*)d_out;

    char* ws = (char*)d_ws;
    const size_t idx_bytes = (size_t)B_ * N1_ * 3 * sizeof(int);               // 384 KB
    const size_t xt_bytes  = (size_t)B_ * N1_ * CIN_ * sizeof(__hip_bfloat16); // 48 MB
    const size_t w_bytes   = (size_t)OUT_ * CIN_ * sizeof(__hip_bfloat16);     // 768 KB
    const size_t y_bytes   = (size_t)B_ * OUT_ * N1_ * sizeof(float);          // 64 MB

    size_t off = 0;
    int*            gidx = (int*)(ws + off);            off += idx_bytes;
    float*          gw   = (float*)(ws + off);          off += idx_bytes;
    __hip_bfloat16* Xhi  = (__hip_bfloat16*)(ws + off); off += xt_bytes;
    __hip_bfloat16* Xlo  = (__hip_bfloat16*)(ws + off); off += xt_bytes;
    __hip_bfloat16* Whi  = (__hip_bfloat16*)(ws + off); off += w_bytes;
    __hip_bfloat16* Wlo  = (__hip_bfloat16*)(ws + off); off += w_bytes;
    float*          Y    = (float*)(ws + off);          off += y_bytes;
    float*          mean = (float*)(ws + off);
    float*          var  = mean + OUT_;

    knn_weights_kernel<<<B_ * (N1_ / 256), 256, 0, stream>>>(xyz1, xyz2, gidx, gw);
    split_w_kernel<<<(OUT_ * CIN_) / 256, 256, 0, stream>>>(Wf, Whi, Wlo);
    build_xt_kernel<<<B_ * (N1_ / 64), 256, 0, stream>>>(feats1, feats2, gidx, gw, Xhi, Xlo);
    gemm_wmma_bf16_kernel<<<dim3(N1_ / BN, OUT_ / BM, B_), 512, 0, stream>>>(
        (const uint32_t*)Whi, (const uint32_t*)Wlo,
        (const uint32_t*)Xhi, (const uint32_t*)Xlo, Y);
    bn_stats_kernel<<<OUT_, 256, 0, stream>>>(Y, mean, var);
    bn_apply_kernel<<<4096, 256, 0, stream>>>(Y, mean, var, gamma, beta, outp);
}
